// RelativeMultiHeadAttentionBlock_24919400251645
// MI455X (gfx1250) — matrix operand order repaired and checked by reading, not compiled, not run on hardware
//
#include <hip/hip_runtime.h>
#include <hip/hip_bf16.h>

// ---- problem constants (match reference) ----
#define BB   8
#define SS   512
#define DD   1024
#define HH   16
#define DK   64
#define MAXL 512

typedef __attribute__((ext_vector_type(16))) __bf16 v16bf;
typedef __attribute__((ext_vector_type(8)))  __bf16 v8bf;
typedef __attribute__((ext_vector_type(8)))  float  v8f;

// D = A(16x32 bf16) * B(32x16 bf16) + C(16x16 f32)
__device__ __forceinline__ v8f wmma_bf16(v16bf a, v16bf b, v8f c) {
  return __builtin_amdgcn_wmma_f32_16x16x32_bf16(
      /*neg_a=*/false, a, /*neg_b=*/false, b,
      /*c_mod=*/(short)0, c, /*reuse_a=*/false, /*reuse_b=*/false);
}

// ---- A-matrix (16x32, 16-bit) per-lane layout ----
// lanes 0-15 : row=lane,    elems 0..15 = K 0..15
// lanes16-31 : row=lane-16, elems 0..15 = K 16..31
__device__ __forceinline__ v16bf load_a_bf16(const __bf16* src, int ld) {
  const int lane = threadIdx.x & 31;
  const int row  = lane & 15;
  const int kb   = (lane >> 4) * 16;
  const __bf16* p = src + (size_t)row * ld;
  v8bf lo = *(const v8bf*)(p + kb);
  v8bf hi = *(const v8bf*)(p + kb + 8);
  v16bf a;
#pragma unroll
  for (int e = 0; e < 8; ++e) { a[e] = lo[e]; a[8 + e] = hi[e]; }
  return a;
}

__device__ __forceinline__ v16bf load_a_f32(const float* src, int ld) {
  const int lane = threadIdx.x & 31;
  const int row  = lane & 15;
  const int kb   = (lane >> 4) * 16;
  const float* p = src + (size_t)row * ld;
  v16bf a;
#pragma unroll
  for (int e = 0; e < 8; ++e) {
    a[e]     = (__bf16)p[kb + e];
    a[8 + e] = (__bf16)p[kb + 8 + e];
  }
  return a;
}

// ---- B-matrix (32x16, 16-bit): lane n<16 -> col n, K 0..15 ; lane>=16 -> col n-16, K 16..31
// source with element (k,n) at src[n*ld + k]   (contiguous in k) -> single 32B vector load
__device__ __forceinline__ v16bf load_b_ncontig(const __bf16* src, int ld) {
  const int lane = threadIdx.x & 31;
  const int n  = lane & 15;
  const int kb = (lane >> 4) * 16;
  return *(const v16bf*)(src + (size_t)n * ld + kb);
}

// ------------------------------------------------------------------
// f32 -> bf16 conversion with zero padding (rel_emb)
__global__ void cvt_f32_bf16(const float* __restrict__ src, __bf16* __restrict__ dst,
                             int n_src, int n_dst) {
  int i = blockIdx.x * blockDim.x + threadIdx.x;
  if (i < n_dst) dst[i] = (i < n_src) ? (__bf16)src[i] : (__bf16)0.0f;
}

// W[1024,1024] f32 -> W^T[1024,1024] bf16  (so GEMM B-tiles load contiguously)
__global__ void cvt_w_transpose(const float* __restrict__ src, __bf16* __restrict__ dst) {
  int idx = blockIdx.x * blockDim.x + threadIdx.x;   // k*1024 + c, read coalesced
  int kk = idx >> 10, c = idx & 1023;
  dst[(size_t)c * DD + kk] = (__bf16)src[idx];
}

// ------------------------------------------------------------------
// Y[b,h,s,dk] (or transposed [b,h,dk,s]) = bf16( X[4096,1024] @ W[1024,1024] )
// One wave computes a 16x64 strip. k-loop is double-buffered AND fully
// unrolled (32 steps) so register renaming removes all buffer-rotation movs.
// grid = (DD/64, B*S/16); WT is W transposed [c][k] bf16.
__global__ void proj_kernel(const float* __restrict__ X, const __bf16* __restrict__ WT,
                            __bf16* __restrict__ Y, int transpose_out) {
  const int lane = threadIdx.x;
  const int c0 = blockIdx.x * 64;
  const int r0 = blockIdx.y * 16;
  const float*  Ab = X + (size_t)r0 * DD;
  const __bf16* Bb = WT + (size_t)c0 * DD;

  v8f acc[4] = {{}, {}, {}, {}};
  v16bf a = load_a_f32(Ab, DD);
  v16bf b[4];
#pragma unroll
  for (int t = 0; t < 4; ++t) b[t] = load_b_ncontig(Bb + (size_t)(t * 16) * DD, DD);

#pragma unroll
  for (int k0 = 0; k0 < DD; k0 += 32) {
    const int kn = k0 + 32;
    v16bf an = a, bn[4] = {b[0], b[1], b[2], b[3]};
    if (kn < DD) {
      __builtin_prefetch(Ab + kn + 32 + (size_t)(lane & 15) * DD + (lane >> 4) * 16);
      __builtin_prefetch(Bb + (size_t)lane * DD + kn + 32);
      __builtin_prefetch(Bb + (size_t)(32 + lane) * DD + kn + 32);
      an = load_a_f32(Ab + kn, DD);
#pragma unroll
      for (int t = 0; t < 4; ++t) bn[t] = load_b_ncontig(Bb + (size_t)(t * 16) * DD + kn, DD);
    }
#pragma unroll
    for (int t = 0; t < 4; ++t) acc[t] = wmma_bf16(a, b[t], acc[t]);
    a = an;
#pragma unroll
    for (int t = 0; t < 4; ++t) b[t] = bn[t];
  }

  const int half = lane >> 4, jn = lane & 15;
#pragma unroll
  for (int t = 0; t < 4; ++t) {
    const int c = c0 + t * 16 + jn, h = c >> 6, dk = c & 63;
#pragma unroll
    for (int r = 0; r < 8; ++r) {
      int row = r0 + r + 8 * half;          // flat b*S + s
      int bb = row >> 9, s = row & 511;
      size_t off = transpose_out
          ? (((size_t)(bb * HH + h)) * DK + dk) * SS + s     // [b,h,dk,s]
          : (((size_t)(bb * HH + h)) * SS + s) * DK + dk;    // [b,h,s,dk]
      Y[off] = (__bf16)acc[t][r];
    }
  }
}

// ------------------------------------------------------------------
// Flash attention with relative position bias.
// grid = (S/16, H, B); one wave (32 threads) per block.
__global__ void attn_kernel(const __bf16* __restrict__ qh, const __bf16* __restrict__ kh,
                            const __bf16* __restrict__ vT, const __bf16* __restrict__ relb,
                            __bf16* __restrict__ aout) {
  __shared__ __bf16 Rb[16][1024];   // R[i', m] = Q_tile . rel_emb[m]   (32 KB)
  __shared__ float  Pt[16][32];     // P tile staging for layout transpose

  const int lane = threadIdx.x;
  const int i0 = blockIdx.x * 16;
  const int h  = blockIdx.y;
  const int b  = blockIdx.z;
  const int half = lane >> 4, jn = lane & 15;

  const __bf16* qbase = qh + (((size_t)b * HH + h) * SS + i0) * DK;
  const __bf16* kbase = kh + (((size_t)b * HH + h) * SS) * DK;
  const __bf16* vbase = vT + (((size_t)b * HH + h) * DK) * SS;

  const v16bf qa0 = load_a_bf16(qbase, DK);        // d = 0..31
  const v16bf qa1 = load_a_bf16(qbase + 32, DK);   // d = 32..63

  // ---- R = Q_tile @ rel_emb^T  ->  LDS (bf16), 64 tiles of 16 columns ----
  for (int mt = 0; mt < 64; ++mt) {
    const int m0 = mt * 16;
    v8f r = {};
    r = wmma_bf16(qa0, load_b_ncontig(relb + (size_t)m0 * DK, DK), r);
    r = wmma_bf16(qa1, load_b_ncontig(relb + (size_t)m0 * DK + 32, DK), r);
#pragma unroll
    for (int rr = 0; rr < 8; ++rr)
      Rb[rr + 8 * half][m0 + jn] = (__bf16)r[rr];
  }
  __syncthreads();

  float mrow[8], lrow[8], scl[8];
  v8f o0 = {}, o1 = {}, o2 = {}, o3 = {};
#pragma unroll
  for (int r = 0; r < 8; ++r) { mrow[r] = -1e30f; lrow[r] = 0.0f; }

  for (int j0 = 0; j0 < SS; j0 += 32) {
    if (j0 + 32 < SS) {  // prefetch next K/V tiles
      __builtin_prefetch(kbase + (size_t)(j0 + 32 + (lane & 31)) * DK);
      __builtin_prefetch(vbase + (size_t)lane * SS + j0 + 32);
      __builtin_prefetch(vbase + (size_t)(lane + 32) * SS + j0 + 32);
    }
    // ---- scores 16x32 = Q @ K^T  (two 16x16 subtiles, K-dim 64 = 2 steps) ----
    v8f s0 = {}, s1 = {};
    s0 = wmma_bf16(qa0, load_b_ncontig(kbase + (size_t)j0 * DK, DK), s0);
    s0 = wmma_bf16(qa1, load_b_ncontig(kbase + (size_t)j0 * DK + 32, DK), s0);
    s1 = wmma_bf16(qa0, load_b_ncontig(kbase + (size_t)(j0 + 16) * DK, DK), s1);
    s1 = wmma_bf16(qa1, load_b_ncontig(kbase + (size_t)(j0 + 16) * DK + 32, DK), s1);

    // ---- V B-tiles hoisted: no dependence on softmax; overlap their latency
    //      with the exp/shuffle/LDS work below (compiler can't hoist past barriers)
    v16bf bv0 = load_b_ncontig(vbase + (size_t)0  * SS + j0, SS);
    v16bf bv1 = load_b_ncontig(vbase + (size_t)16 * SS + j0, SS);
    v16bf bv2 = load_b_ncontig(vbase + (size_t)32 * SS + j0, SS);
    v16bf bv3 = load_b_ncontig(vbase + (size_t)48 * SS + j0, SS);

    // ---- relative bias gather + scale (1/sqrt(64)) ----
#pragma unroll
    for (int r = 0; r < 8; ++r) {
      const int iq = r + 8 * half;                    // local row
      const int m = j0 + jn - (i0 + iq) + (MAXL - 1); // in [0,1022]
      s0[r] = (s0[r] + (float)Rb[iq][m]) * 0.125f;
      s1[r] = (s1[r] + (float)Rb[iq][m + 16]) * 0.125f;
    }

    // ---- online softmax (row reductions across 16-lane halves) ----
#pragma unroll
    for (int r = 0; r < 8; ++r) {
      float mx = fmaxf(s0[r], s1[r]);
      for (int d = 8; d >= 1; d >>= 1) mx = fmaxf(mx, __shfl_xor(mx, d, 32));
      const float mn = fmaxf(mrow[r], mx);
      scl[r] = __expf(mrow[r] - mn);
      s0[r] = __expf(s0[r] - mn);
      s1[r] = __expf(s1[r] - mn);
      float sum = s0[r] + s1[r];
      for (int d = 8; d >= 1; d >>= 1) sum += __shfl_xor(sum, d, 32);
      lrow[r] = lrow[r] * scl[r] + sum;
      mrow[r] = mn;
      o0[r] *= scl[r]; o1[r] *= scl[r]; o2[r] *= scl[r]; o3[r] *= scl[r];
    }

    // ---- P (C layout) -> LDS -> A layout, bf16 ----
#pragma unroll
    for (int r = 0; r < 8; ++r) {
      Pt[r + 8 * half][jn]      = s0[r];
      Pt[r + 8 * half][16 + jn] = s1[r];
    }
    __syncthreads();
    v16bf pa;
    {
      const int row = lane & 15, kb = (lane >> 4) * 16;
#pragma unroll
      for (int e = 0; e < 8; ++e) {
        pa[e]     = (__bf16)Pt[row][kb + e];
        pa[8 + e] = (__bf16)Pt[row][kb + 8 + e];
      }
    }
    __syncthreads();

    // ---- O += P(16x32) @ V(32x64)  : 4 n-tiles, single K step ----
    o0 = wmma_bf16(pa, bv0, o0);
    o1 = wmma_bf16(pa, bv1, o1);
    o2 = wmma_bf16(pa, bv2, o2);
    o3 = wmma_bf16(pa, bv3, o3);
  }

  // ---- normalize and write attn_out[b, s, h*64+dk] (bf16) ----
#pragma unroll
  for (int r = 0; r < 8; ++r) {
    const int iq = i0 + r + 8 * half;
    const float inv = 1.0f / lrow[r];
    const size_t base = ((size_t)b * SS + iq) * DD + (size_t)h * DK;
    aout[base + jn]      = (__bf16)(o0[r] * inv);
    aout[base + 16 + jn] = (__bf16)(o1[r] * inv);
    aout[base + 32 + jn] = (__bf16)(o2[r] * inv);
    aout[base + 48 + jn] = (__bf16)(o3[r] * inv);
  }
}

// ------------------------------------------------------------------
// d_out[4096,1024] (f32) = attn_out(bf16) @ Wo (via WoT bf16); 16x64 strip per wave,
// double-buffered + fully unrolled like proj_kernel.
__global__ void out_gemm(const __bf16* __restrict__ A, const __bf16* __restrict__ WT,
                         float* __restrict__ Y) {
  const int lane = threadIdx.x;
  const int c0 = blockIdx.x * 64;
  const int r0 = blockIdx.y * 16;
  const __bf16* Ab = A + (size_t)r0 * DD;
  const __bf16* Bb = WT + (size_t)c0 * DD;

  v8f acc[4] = {{}, {}, {}, {}};
  v16bf a = load_a_bf16(Ab, DD);
  v16bf b[4];
#pragma unroll
  for (int t = 0; t < 4; ++t) b[t] = load_b_ncontig(Bb + (size_t)(t * 16) * DD, DD);

#pragma unroll
  for (int k0 = 0; k0 < DD; k0 += 32) {
    const int kn = k0 + 32;
    v16bf an = a, bn[4] = {b[0], b[1], b[2], b[3]};
    if (kn < DD) {
      __builtin_prefetch(Ab + (size_t)(lane & 15) * DD + kn + 32);
      __builtin_prefetch(Bb + (size_t)lane * DD + kn + 32);
      __builtin_prefetch(Bb + (size_t)(32 + lane) * DD + kn + 32);
      an = load_a_bf16(Ab + kn, DD);
#pragma unroll
      for (int t = 0; t < 4; ++t) bn[t] = load_b_ncontig(Bb + (size_t)(t * 16) * DD + kn, DD);
    }
#pragma unroll
    for (int t = 0; t < 4; ++t) acc[t] = wmma_bf16(a, b[t], acc[t]);
    a = an;
#pragma unroll
    for (int t = 0; t < 4; ++t) b[t] = bn[t];
  }

  const int half = lane >> 4, jn = lane & 15;
#pragma unroll
  for (int t = 0; t < 4; ++t)
#pragma unroll
    for (int r = 0; r < 8; ++r)
      Y[(size_t)(r0 + r + 8 * half) * DD + c0 + t * 16 + jn] = acc[t][r];
}

// ------------------------------------------------------------------
extern "C" void kernel_launch(void* const* d_in, const int* in_sizes, int n_in,
                              void* d_out, int out_size, void* d_ws, size_t ws_size,
                              hipStream_t stream) {
  const float* q   = (const float*)d_in[0];
  const float* k   = (const float*)d_in[1];
  const float* v   = (const float*)d_in[2];
  const float* Wq  = (const float*)d_in[3];
  const float* Wk  = (const float*)d_in[4];
  const float* Wv  = (const float*)d_in[5];
  const float* Wo  = (const float*)d_in[6];
  const float* rel = (const float*)d_in[7];
  float* out = (float*)d_out;

  char* ws = (char*)d_ws;
  const size_t MB = 1024 * 1024;
  __bf16* WqT  = (__bf16*)(ws + 0  * MB);   // transposed bf16 weights, 2 MB each
  __bf16* WkT  = (__bf16*)(ws + 2  * MB);
  __bf16* WvT  = (__bf16*)(ws + 4  * MB);
  __bf16* WoT  = (__bf16*)(ws + 6  * MB);
  __bf16* relb = (__bf16*)(ws + 8  * MB);   // 1024x64 bf16 (padded), 128 KB
  __bf16* qhb  = (__bf16*)(ws + 10 * MB);   // [B,H,S,64]  8 MB
  __bf16* khb  = (__bf16*)(ws + 18 * MB);   // [B,H,S,64]  8 MB
  __bf16* vTb  = (__bf16*)(ws + 26 * MB);   // [B,H,64,S]  8 MB
  __bf16* aob  = (__bf16*)(ws + 34 * MB);   // [B*S, D]    8 MB

  const int nW = DD * DD;
  cvt_w_transpose<<<nW / 256, 256, 0, stream>>>(Wq, WqT);
  cvt_w_transpose<<<nW / 256, 256, 0, stream>>>(Wk, WkT);
  cvt_w_transpose<<<nW / 256, 256, 0, stream>>>(Wv, WvT);
  cvt_w_transpose<<<nW / 256, 256, 0, stream>>>(Wo, WoT);
  cvt_f32_bf16<<<(1024 * DK + 255) / 256, 256, 0, stream>>>(
      rel, relb, (2 * MAXL - 1) * DK, 1024 * DK);

  dim3 gG(DD / 64, (BB * SS) / 16);           // 16 x 256 strips
  proj_kernel<<<gG, 32, 0, stream>>>(q, WqT, qhb, 0);
  proj_kernel<<<gG, 32, 0, stream>>>(k, WkT, khb, 0);
  proj_kernel<<<gG, 32, 0, stream>>>(v, WvT, vTb, 1);

  dim3 gA(SS / 16, HH, BB);                   // 32 x 16 x 8 blocks, 1 wave each
  attn_kernel<<<gA, 32, 0, stream>>>(qhb, khb, vTb, relb, aob);

  out_gemm<<<gG, 32, 0, stream>>>(aob, WoT, out);
}
